// NCA_67654324846917
// MI455X (gfx1250) — matrix operand order repaired and checked by reading, not compiled, not run on hardware
//
#include <hip/hip_runtime.h>

typedef _Float16 f16;
typedef __attribute__((ext_vector_type(2)))  _Float16 h2;
typedef __attribute__((ext_vector_type(16))) _Float16 v16h;
typedef __attribute__((ext_vector_type(8)))  _Float16 v8h;
typedef __attribute__((ext_vector_type(8)))  float    v8f;
typedef __attribute__((ext_vector_type(4)))  float    v4f;

#define CBAR() asm volatile("" ::: "memory")

// Problem sizes (fixed by the reference)
#define NB 16
#define NC 16
#define NH 256
#define NW 256

// LDS pitches (f16 units) chosen so lane n's 16B reads hit distinct banks:
//  perc pitch 72  -> byte pitch 144, bank offset n*36 mod 64 : all distinct
//  hstg pitch 136 -> byte pitch 272, bank offset n*4         : all distinct
#define PPITCH 72
#define HPITCH 136

// Prepacked weight image: 20 fragments (16 = W1[b1-folded], 4 = W2),
// each fragment = 32 lanes x 16 f16 (1 KB), total 20 KB.
#define WPK_BYTES (20 * 1024)

// ---- pack W1 (+b1 in K=48 col) and W2 into per-lane WMMA A-fragment images ----
__global__ void pack_weights_kernel(const float* __restrict__ W1,
                                    const float* __restrict__ b1,
                                    const float* __restrict__ W2,
                                    f16* __restrict__ wpk)
{
    const int idx = blockIdx.x * 256 + threadIdx.x;   // 5120 f16-pairs
    if (idx >= 5120) return;
    const int f    = idx >> 8;        // fragment 0..19
    const int rem  = idx & 255;
    const int l    = rem >> 3;        // lane 0..31
    const int v    = rem & 7;         // pair slot 0..7
    const int half = l >> 4;
    const int lm   = l & 15;
    // A-matrix 16x32 f16 layout (ISA 7.12.2): VGPR v holds K = (v/4)*16 + half*8 + (v%4)*2
    int k = (v >> 2) * 16 + half * 8 + (v & 3) * 2;
    float a0 = 0.f, a1 = 0.f;
    if (f < 16) {                     // W1 fragment (mt, kf)
        const int mt = f >> 1, kf = f & 1;
        const int o  = mt * 16 + lm;
        k += kf * 32;
        if (k < 48)       { a0 = W1[o * 48 + k]; a1 = W1[o * 48 + k + 1]; }
        else if (k == 48) { a0 = b1[o]; }
    } else {                          // W2 fragment kf
        const int kf = f - 16;
        k += kf * 32;
        a0 = W2[lm * 128 + k];
        a1 = W2[lm * 128 + k + 1];
    }
    h2 p; p[0] = (f16)a0; p[1] = (f16)a1;
    *(h2*)&wpk[f * 512 + l * 16 + 2 * v] = p;
}

template <bool PREPACKED>
__launch_bounds__(64)
__global__ void nca_fused_kernel(const float* __restrict__ grid,
                                 const float* __restrict__ W1,
                                 const float* __restrict__ b1,
                                 const float* __restrict__ W2,
                                 const float* __restrict__ b2,
                                 const f16* __restrict__ wpk,
                                 float* __restrict__ out)
{
    // rows2[ch2][r][x] : one h2 = channels (2*ch2, 2*ch2+1) at (row r, col x)
    __shared__ h2  rows2[8 * 3 * NW];        // 24 KB
    __shared__ f16 perc[2][16 * PPITCH];     // per-wave perception staging (48 ch + bias col + pad)
    __shared__ f16 hstg[2][16 * HPITCH];     // per-wave hidden staging: 16 px x 128 ch

    const int tid  = threadIdx.x;
    const int lane = tid & 31;
    const int wave = tid >> 5;
    const int half = lane >> 4;   // 0: lanes 0-15, 1: lanes 16-31
    const int lm   = lane & 15;

    const int b = blockIdx.x >> 8;    // batch
    const int h = blockIdx.x & 255;   // row

    // ------- load 3 input rows, channel-pair packed f32 -> h2, into LDS -------
    {
        const int x0 = tid * 4;  // 64 threads x 4 cols = 256 = full row
        #pragma unroll 3
        for (int cr = 0; cr < 24; ++cr) {
            const int ch2 = cr / 3;
            const int r   = cr % 3;
            const int hr  = h - 1 + r;
            v4f a = {}, c = {};
            if (hr >= 0 && hr < NH) {
                const float* base = grid + ((b * NC) * NH + hr) * NW + x0;
                a = *(const v4f*)(base + (2 * ch2)     * NH * NW);
                c = *(const v4f*)(base + (2 * ch2 + 1) * NH * NW);
            }
            v8h s;   // interleaved: [chA,chB] per column
            s[0] = (f16)a[0]; s[1] = (f16)c[0];
            s[2] = (f16)a[1]; s[3] = (f16)c[1];
            s[4] = (f16)a[2]; s[5] = (f16)c[2];
            s[6] = (f16)a[3]; s[7] = (f16)c[3];
            *(v8h*)&rows2[(ch2 * 3 + r) * NW + x0] = s;
        }
    }

    // ---------------- weight fragments in registers --------------------------
    v16h w1f[8][2];   // 8 out-ch tiles x 2 K-frags (K padded 48->64, col 48 = b1)
    v16h w2f[4];      // 16 out x K=128 -> 4 K-frags
    if constexpr (PREPACKED) {
        const v16h* wv = (const v16h*)wpk;   // frag f, lane l at index f*32 + l (32B aligned)
        #pragma unroll
        for (int mt = 0; mt < 8; ++mt) {
            w1f[mt][0] = wv[(mt * 2 + 0) * 32 + lane];
            w1f[mt][1] = wv[(mt * 2 + 1) * 32 + lane];
        }
        #pragma unroll
        for (int kf = 0; kf < 4; ++kf) w2f[kf] = wv[(16 + kf) * 32 + lane];
    } else {
        #pragma unroll
        for (int mt = 0; mt < 8; ++mt) {
            const int o = mt * 16 + lm;
            #pragma unroll
            for (int kf = 0; kf < 2; ++kf) {
                #pragma unroll
                for (int v = 0; v < 8; ++v) {
                    const int k = kf * 32 + (v >> 2) * 16 + half * 8 + (v & 3) * 2;
                    float a0 = 0.f, a1 = 0.f;
                    if (k < 48)       { a0 = W1[o * 48 + k]; a1 = W1[o * 48 + k + 1]; }
                    else if (k == 48) { a0 = b1[o]; }
                    w1f[mt][kf][2 * v]     = (f16)a0;
                    w1f[mt][kf][2 * v + 1] = (f16)a1;
                }
            }
        }
        #pragma unroll
        for (int kf = 0; kf < 4; ++kf) {
            #pragma unroll
            for (int v = 0; v < 8; ++v) {
                const int k = kf * 32 + (v >> 2) * 16 + half * 8 + (v & 3) * 2;
                w2f[kf][2 * v]     = (f16)W2[lm * 128 + k];
                w2f[kf][2 * v + 1] = (f16)W2[lm * 128 + k + 1];
            }
        }
    }
    // C/D f32 layout: lane = 16*half + n, VGPR r holds M = r + 8*half
    float b2r[8];
    #pragma unroll
    for (int r = 0; r < 8; ++r) b2r[r] = b2[half * 8 + r];

    // K=48..63 region of perception staging: col 48 = 1.0 (bias), rest 0.
    // Written once; the tile loop only ever writes cols 0..47.
    {
        v8h z;
        #pragma unroll
        for (int e = 0; e < 8; ++e) z[e] = (f16)0.f;
        if (half == 0) z[0] = (f16)1.0f;
        *(v8h*)&perc[wave][lm * PPITCH + 48 + half * 8] = z;
    }

    __syncthreads();   // rows2[] visible to both waves

    const h2 two    = {(f16)2.0f, (f16)2.0f};
    const h2 eighth = {(f16)0.125f, (f16)0.125f};
    const h2 zero2  = {(f16)0.0f, (f16)0.0f};

    // ---------------- per-tile loop: 16 pixels per iteration ------------------
    for (int t = wave; t < 16; t += 2) {
        const int x0 = t * 16;

        // ---- perception (packed 2 channels/lane): grid | sobel_x | sobel_y ----
        #pragma unroll
        for (int u = 0; u < 4; ++u) {
            const int idx = u * 32 + lane;
            const int px  = idx & 15;     // pixel in tile
            const int ch2 = idx >> 4;     // channel pair 0..7
            const int x   = x0 + px;
            const h2* rp  = &rows2[ch2 * 3 * NW];

            const h2 c0 = rp[0 * NW + x];
            const h2 c1 = rp[1 * NW + x];
            const h2 c2 = rp[2 * NW + x];
            h2 l0 = zero2, l1 = zero2, l2 = zero2;
            h2 r0 = zero2, r1 = zero2, r2 = zero2;
            if (x > 0)      { l0 = rp[0*NW + x-1]; l1 = rp[1*NW + x-1]; l2 = rp[2*NW + x-1]; }
            if (x < NW - 1) { r0 = rp[0*NW + x+1]; r1 = rp[1*NW + x+1]; r2 = rp[2*NW + x+1]; }

            const h2 sx = ((r0 - l0) + two * (r1 - l1) + (r2 - l2)) * eighth;
            const h2 sy = ((l2 - l0) + two * (c2 - c0) + (r2 - r0)) * eighth;

            h2* pp = (h2*)&perc[wave][px * PPITCH + 2 * ch2];
            pp[0]  = c1;   // identity channels 0..15
            pp[8]  = sx;   // sobel_x channels 16..31
            pp[16] = sy;   // sobel_y channels 32..47
        }
        CBAR();   // LDS is in-order within a wave; just block IR reordering

        // ---- GEMM1: h = relu(W1 * perception [+ b1 via K=48 col]) ----
        // B-matrix layout: lane = 16*half + n; VGPR v holds K = (v/4)*16 + half*8 + 2(v%4) (+1)
        v16h pb[2];
        #pragma unroll
        for (int kf = 0; kf < 2; ++kf) {
            const f16* p = &perc[wave][lm * PPITCH + kf * 32 + half * 8];
            const v8h lo = *(const v8h*)p;
            const v8h hi = *(const v8h*)(p + 16);
            #pragma unroll
            for (int e = 0; e < 8; ++e) { pb[kf][e] = lo[e]; pb[kf][8 + e] = hi[e]; }
        }
        #pragma unroll
        for (int mt = 0; mt < 8; ++mt) {
            v8f c = {};
            c = __builtin_amdgcn_wmma_f32_16x16x32_f16(false, w1f[mt][0], false, pb[0], (short)0, c, false, false);
            c = __builtin_amdgcn_wmma_f32_16x16x32_f16(false, w1f[mt][1], false, pb[1], (short)0, c, false, false);
            // cvt to f16 pairs first, then packed ReLU (v_cvt_pk + v_pk_max)
            v8h hv;
            #pragma unroll
            for (int r2 = 0; r2 < 4; ++r2) {
                h2 p;
                p[0] = (f16)c[2 * r2];
                p[1] = (f16)c[2 * r2 + 1];
                p = __builtin_elementwise_max(p, zero2);
                hv[2 * r2]     = p[0];
                hv[2 * r2 + 1] = p[1];
            }
            *(v8h*)&hstg[wave][lm * HPITCH + mt * 16 + half * 8] = hv;
        }
        CBAR();

        // ---- GEMM2: out = W2 * h + b2 ----
        v8f c2 = {};
        #pragma unroll
        for (int kf = 0; kf < 4; ++kf) {
            const f16* p = &hstg[wave][lm * HPITCH + kf * 32 + half * 8];
            const v8h lo = *(const v8h*)p;
            const v8h hi = *(const v8h*)(p + 16);
            v16h hb;
            #pragma unroll
            for (int e = 0; e < 8; ++e) { hb[e] = lo[e]; hb[8 + e] = hi[e]; }
            c2 = __builtin_amdgcn_wmma_f32_16x16x32_f16(false, w2f[kf], false, hb, (short)0, c2, false, false);
        }
        #pragma unroll
        for (int r = 0; r < 8; ++r) {
            const int m = r + 8 * half;
            out[(((b * NC + m) * NH) + h) * NW + x0 + lm] = c2[r] + b2r[r];
        }
    }
}

extern "C" void kernel_launch(void* const* d_in, const int* in_sizes, int n_in,
                              void* d_out, int out_size, void* d_ws, size_t ws_size,
                              hipStream_t stream) {
    const float* grid = (const float*)d_in[0];
    const float* W1   = (const float*)d_in[1];
    const float* b1   = (const float*)d_in[2];
    const float* W2   = (const float*)d_in[3];
    const float* b2   = (const float*)d_in[4];
    float* out = (float*)d_out;

    if (ws_size >= (size_t)WPK_BYTES && d_ws != nullptr) {
        f16* wpk = (f16*)d_ws;
        pack_weights_kernel<<<dim3(20), dim3(256), 0, stream>>>(W1, b1, W2, wpk);
        nca_fused_kernel<true><<<dim3(NB * NH), dim3(64), 0, stream>>>(
            grid, W1, b1, W2, b2, wpk, out);
    } else {
        nca_fused_kernel<false><<<dim3(NB * NH), dim3(64), 0, stream>>>(
            grid, W1, b1, W2, b2, nullptr, out);
    }
}
